// TopKGating_51144470560937
// MI455X (gfx1250) — matrix operand order repaired and checked by reading, not compiled, not run on hardware
//
#include <hip/hip_runtime.h>
#include <hip/hip_bf16.h>
#include <math.h>

typedef __attribute__((ext_vector_type(16))) _Float16 v16h;
typedef __attribute__((ext_vector_type(8)))  _Float16 v8h;
typedef __attribute__((ext_vector_type(8)))  float    v8f;
typedef __attribute__((ext_vector_type(4)))  float    v4f;

#define NUM_EXPERTS 64
#define INPUT_DIM   1024
#define KCHUNKS1    (INPUT_DIM / 32)    // 32 chunks of K=32 for GEMM1 f16 WMMA
#define KCHUNKS2    (NUM_EXPERTS / 32)  // 2  chunks of K=32 for GEMM2 f16 WMMA
#define ETILES      4                   // 64 experts / 16
#define LDS_HSTRIDE 72                  // padded f16 row stride (halves)

// ---------------------------------------------------------------------------
// Pack kernel: rearrange W_gate (f32 [K=1024, E=64]) and w_noise (f32 [64,64])
// into f16 WMMA B-fragments in d_ws.
//
// f16 B fragment (32x16): lanes 0-15 hold K 0..7,16..23; lanes 16-31 hold
// K 8..15,24..31  =>  half h maps to k = 8*hi+h (h<8) / 16+8*hi+(h-8).
// ---------------------------------------------------------------------------
__global__ void moe_pack_kernel(const float* __restrict__ W_gate,
                                const float* __restrict__ w_noise,
                                _Float16* __restrict__ wpack,
                                _Float16* __restrict__ npack) {
  int tid = blockIdx.x * blockDim.x + threadIdx.x;
  int nthreads = gridDim.x * blockDim.x;
  // wpack[chunk][tile][lane][h]  : 32*4*32*16 halves = 128 KB
  for (int i = tid; i < KCHUNKS1 * ETILES * 32 * 16; i += nthreads) {
    int h    = i & 15;
    int lane = (i >> 4) & 31;
    int t    = (i >> 9) & 3;
    int c    = i >> 11;
    int k    = c * 32 + ((h < 8) ? ((lane >> 4) * 8 + h) : (16 + (lane >> 4) * 8 + (h - 8)));
    int e    = t * 16 + (lane & 15);
    wpack[i] = (_Float16)W_gate[k * NUM_EXPERTS + e];
  }
  // npack[chunk][tile][lane][h]  : 2*4*32*16 halves = 8 KB
  for (int i = tid; i < KCHUNKS2 * ETILES * 32 * 16; i += nthreads) {
    int h    = i & 15;
    int lane = (i >> 4) & 31;
    int t    = (i >> 9) & 3;
    int c    = i >> 11;
    int k    = c * 32 + ((h < 8) ? ((lane >> 4) * 8 + h) : (16 + (lane >> 4) * 8 + (h - 8)));
    int e    = t * 16 + (lane & 15);
    npack[i] = (_Float16)w_noise[k * NUM_EXPERTS + e];
  }
}

// ---------------------------------------------------------------------------
// Fused gating kernel. One wave per 16-token x 64-expert tile.
//   GEMM1: clean = x @ W_gate + b   (v_wmma_f32_16x16x32_f16, f32 accum)
//   GEMM2: raw   = clean @ w_noise  (clean cast to f16 via LDS, same WMMA)
//   then softplus-noise, top-2 threshold, double softmax, store gates.
// ---------------------------------------------------------------------------
__global__ void __launch_bounds__(256)
moe_gate_kernel(const float* __restrict__ x,
                const float* __restrict__ b_gate,
                const float* __restrict__ noise,
                const _Float16* __restrict__ wpack,
                const _Float16* __restrict__ npack,
                float* __restrict__ gates,
                int n_tiles) {
  __shared__ _Float16 lds_clean[8][16][LDS_HSTRIDE];  // per-wave 16x64 f16 tile

  const int lane   = threadIdx.x & 31;
  const int wave   = threadIdx.x >> 5;
  const int lane16 = lane & 15;
  const int hi     = lane >> 4;                       // which 16-lane half
  _Float16 (*clh)[LDS_HSTRIDE] = lds_clean[wave];

  const int wavesTotal = (gridDim.x * blockDim.x) >> 5;
  const int waveId     = (blockIdx.x * blockDim.x + threadIdx.x) >> 5;

  for (int tile = waveId; tile < n_tiles; tile += wavesTotal) {
    const int row0 = tile << 4;

    // ---- init accumulators with bias (C: row m=j+8*hi, col = 16t+lane16) ----
    v8f acc[ETILES];
#pragma unroll
    for (int t = 0; t < ETILES; ++t) {
      float b = b_gate[t * 16 + lane16];
#pragma unroll
      for (int j = 0; j < 8; ++j) acc[t][j] = b;
    }

    // ---- GEMM1: stream x row tile, cvt f32->f16, 4 WMMA per K-chunk ----
    // A 16x32 f16: lane<16: row=lane16, k = h (h<8) / h+8 (h>=8)
    //              lane>=16: row=lane16, k = h+8 / h+16
    const float* xb = x + (size_t)(row0 + lane16) * INPUT_DIM + (hi ? 8 : 0);
    const v16h* wp = (const v16h*)wpack;
#pragma unroll 4
    for (int c = 0; c < KCHUNKS1; ++c) {
      const float* p = xb + c * 32;
      v4f a0 = *(const v4f*)(p);
      v4f a1 = *(const v4f*)(p + 4);
      v4f a2 = *(const v4f*)(p + 16);
      v4f a3 = *(const v4f*)(p + 20);
      v16h A;
#pragma unroll
      for (int q = 0; q < 4; ++q) {
        A[q]      = (_Float16)a0[q];
        A[4 + q]  = (_Float16)a1[q];
        A[8 + q]  = (_Float16)a2[q];
        A[12 + q] = (_Float16)a3[q];
      }
      const v16h* bp = wp + (c * ETILES) * 32 + lane;
#pragma unroll
      for (int t = 0; t < ETILES; ++t) {
        v16h B = bp[t * 32];
        acc[t] = __builtin_amdgcn_wmma_f32_16x16x32_f16(
            false, A, false, B, (short)0, acc[t], false, false);
      }
    }

    // ---- stage clean tile to LDS as f16 (row-major, padded stride) ----
#pragma unroll
    for (int t = 0; t < ETILES; ++t)
#pragma unroll
      for (int j = 0; j < 8; ++j)
        clh[j + hi * 8][t * 16 + lane16] = (_Float16)acc[t][j];

    asm volatile("s_wait_dscnt 0" ::: "memory");

    // ---- GEMM2: raw = clean_f16 @ w_noise_f16, 2 K-chunks x 4 tiles ----
    v8f racc[ETILES];
#pragma unroll
    for (int t = 0; t < ETILES; ++t)
#pragma unroll
      for (int j = 0; j < 8; ++j) racc[t][j] = 0.f;

    const v16h* np = (const v16h*)npack;
#pragma unroll
    for (int c = 0; c < KCHUNKS2; ++c) {
      const _Float16* arow = &clh[lane16][c * 32 + hi * 8];
      v8h lo  = *(const v8h*)(arow);        // k = base + 0..7
      v8h hi8 = *(const v8h*)(arow + 16);   // k = base + 16..23
      v16h A;
#pragma unroll
      for (int q = 0; q < 8; ++q) { A[q] = lo[q]; A[8 + q] = hi8[q]; }
#pragma unroll
      for (int t = 0; t < ETILES; ++t) {
        v16h B = np[(c * ETILES + t) * 32 + lane];
        racc[t] = __builtin_amdgcn_wmma_f32_16x16x32_f16(
            false, A, false, B, (short)0, racc[t], false, false);
      }
    }

    // ---- elementwise: noise std, logits, top-2, softmax, transform ----
    const float* nz = noise + (size_t)row0 * NUM_EXPERTS;
    const int mbase = hi << 3;
#pragma unroll
    for (int j = 0; j < 8; ++j) {
      const int r = mbase + j;
      float l[ETILES];
#pragma unroll
      for (int t = 0; t < ETILES; ++t) {
        float raw = racc[t][j];
        // softplus via hardware trans ops (guard large args)
        float sp  = (raw > 15.f) ? raw : __logf(1.f + __expf(raw));
        float sd  = sp + 1e-5f;
        float nv  = nz[r * NUM_EXPERTS + t * 16 + lane16];
        l[t] = acc[t][j] + nv * sd;
      }
      // per-lane top-2 over the 4 local values
      float mx01 = fmaxf(l[0], l[1]), mn01 = fminf(l[0], l[1]);
      float mx23 = fmaxf(l[2], l[3]), mn23 = fminf(l[2], l[3]);
      float a1 = fmaxf(mx01, mx23);
      float a2 = fmaxf(fminf(mx01, mx23), fmaxf(mn01, mn23));
      // butterfly top-2 merge across the 16-lane half (masks < 16 stay in half)
#pragma unroll
      for (int s = 1; s < 16; s <<= 1) {
        float b1 = __shfl_xor(a1, s);
        float b2 = __shfl_xor(a2, s);
        float n1 = fmaxf(a1, b1);
        float n2 = fmaxf(fminf(a1, b1), fmaxf(a2, b2));
        a1 = n1; a2 = n2;
      }
      // stable softmax over the row, then masked log1p/expm1 transform
      float e[ETILES], sum = 0.f;
#pragma unroll
      for (int t = 0; t < ETILES; ++t) { e[t] = __expf(l[t] - a1); sum += e[t]; }
#pragma unroll
      for (int s = 1; s < 16; s <<= 1) sum += __shfl_xor(sum, s);
      float inv = __builtin_amdgcn_rcpf(sum);
#pragma unroll
      for (int t = 0; t < ETILES; ++t) {
        float sft = e[t] * inv;   // in (0,1)
        acc[t][j] = (l[t] < a2) ? 10.f * __logf(1.f + sft)
                                : 10.f * (__expf(sft) - 1.f);
      }
    }

    // ---- second softmax + store gates ----
    float* op0 = gates + (size_t)row0 * NUM_EXPERTS;
#pragma unroll
    for (int j = 0; j < 8; ++j) {
      const int r = mbase + j;
      float mx = fmaxf(fmaxf(acc[0][j], acc[1][j]), fmaxf(acc[2][j], acc[3][j]));
#pragma unroll
      for (int s = 1; s < 16; s <<= 1) mx = fmaxf(mx, __shfl_xor(mx, s));
      float e[ETILES], sum = 0.f;
#pragma unroll
      for (int t = 0; t < ETILES; ++t) { e[t] = __expf(acc[t][j] - mx); sum += e[t]; }
#pragma unroll
      for (int s = 1; s < 16; s <<= 1) sum += __shfl_xor(sum, s);
      float inv = __builtin_amdgcn_rcpf(sum);
      float* op = op0 + r * NUM_EXPERTS + lane16;
#pragma unroll
      for (int t = 0; t < ETILES; ++t) op[t * 16] = e[t] * inv;
    }
  }
}

extern "C" void kernel_launch(void* const* d_in, const int* in_sizes, int n_in,
                              void* d_out, int out_size, void* d_ws, size_t ws_size,
                              hipStream_t stream) {
  const float* x       = (const float*)d_in[0];
  const float* W_gate  = (const float*)d_in[1];
  const float* b_gate  = (const float*)d_in[2];
  const float* w_noise = (const float*)d_in[3];
  const float* noise   = (const float*)d_in[4];
  float* gates = (float*)d_out;

  const int N = in_sizes[4] / NUM_EXPERTS;   // noise is [N, 64]
  const int n_tiles = N >> 4;

  // ws layout: 128 KB f16 W_gate fragments, then 8 KB f16 w_noise fragments
  _Float16* wpack = (_Float16*)d_ws;
  _Float16* npack = (_Float16*)((char*)d_ws +
                    (size_t)KCHUNKS1 * ETILES * 32 * 16 * sizeof(_Float16));

  moe_pack_kernel<<<64, 256, 0, stream>>>(W_gate, w_noise, wpack, npack);

  int blocks = (n_tiles + 7) / 8;            // 8 waves per block, 1 tile per wave
  if (blocks > 2048) blocks = 2048;          // grid-stride covers the rest
  if (blocks < 1) blocks = 1;
  moe_gate_kernel<<<blocks, 256, 0, stream>>>(x, b_gate, noise, wpack, npack, gates, n_tiles);
}
